// FlaxDeepseekV2DecoderLayer_12721693130965
// MI455X (gfx1250) — compile-verified, matrix operand order checked by reading, not run on hardware
//
#include <hip/hip_runtime.h>
#include <hip/hip_bf16.h>
#include <math.h>

// ---------------------------------------------------------------------------
// DeepSeek-V2 decoder layer for MI455X (gfx1250, wave32, WMMA bf16)
// ---------------------------------------------------------------------------
typedef __bf16 bf16_t;
typedef __attribute__((ext_vector_type(16))) __bf16 v16bf;
typedef __attribute__((ext_vector_type(8)))  float  v8f;

#define H16     16
#define DN      128     // d_nope
#define DR      64      // d_rope
#define DQ      192     // q head dim
#define DVV     128     // v head dim
#define HIDDEN  2048
#define QLORA_  1536
#define KVL     512
#define NEXP    8
#define INTER_  1408
#define SINTER  2816
#define SEQ     2048
#define BATCH   2
#define NTOK    (BATCH * SEQ)

static __device__ __forceinline__ v8f wmma_bf16(v16bf a, v16bf b, v8f c) {
  return __builtin_amdgcn_wmma_f32_16x16x32_bf16(false, a, false, b, (short)0, c,
                                                 false, false);
}

// Fragment loader for the CDNA5 16x16x32 bf16 layout. Works for A fragments
// (rows = M) and B fragments (rows = N, tile stored N-major) since the layouts
// mirror each other. lanes 0-15: row = lane, K chunks {0..7, 16..23};
// lanes 16-31: row = lane-16, K chunks {8..15, 24..31}. Two 16B loads.
static __device__ __forceinline__ v16bf load_frag(const bf16_t* base, int stride,
                                                  int row16, int k0, int lane) {
  union { v16bf v; float4 q[2]; } u;
  const int r  = row16 + (lane & 15);
  const int c0 = k0 + ((lane >> 4) << 3);
  const bf16_t* p = base + r * stride + c0;
  u.q[0] = *(const float4*)(p);
  u.q[1] = *(const float4*)(p + 16);
  return u.v;
}

// ---------------------------------------------------------------------------
// Generic bf16 WMMA GEMM: C[M,N](f32) = A[M,K](bf16) x B[K,N](bf16)
// mode 0: C = acc
// mode 1: C = acc + resid           (residual add epilogue)
// mode 2: C += rowscale[m]*acc      (router-weighted accumulate, tile skip)
// ---------------------------------------------------------------------------
#define GBM 128
#define GBN 128
#define GBK 32
#define GLD 40   // padded LDS row stride (bf16), 80B rows -> 16B aligned

__global__ __launch_bounds__(256) void gemm_bf16_k(
    const bf16_t* __restrict__ A, const bf16_t* __restrict__ B,
    float* __restrict__ C, const float* __restrict__ resid,
    const float* __restrict__ rowscale, int rs_stride,
    int M, int N, int K, int mode)
{
  __shared__ alignas(16) bf16_t As[GBM * GLD];
  __shared__ alignas(16) bf16_t Bs[GBN * GLD];
  __shared__ int s_live;

  const int tid  = threadIdx.x;
  const int lane = tid & 31;
  const int wave = tid >> 5;
  const int m0 = blockIdx.y * GBM;
  const int n0 = blockIdx.x * GBN;

  if (mode == 2) {                       // skip tiles with no routed tokens
    if (tid == 0) s_live = 0;
    __syncthreads();
    if (tid < GBM && rowscale[(size_t)(m0 + tid) * rs_stride] != 0.0f) s_live = 1;
    __syncthreads();
    if (!s_live) return;
  }

  const int wm = (wave & 3) * 32;        // 4 wave rows x 32
  const int wn = (wave >> 2) * 64;       // 2 wave cols x 64

  v8f acc[2][4];
#pragma unroll
  for (int i = 0; i < 2; ++i)
#pragma unroll
    for (int j = 0; j < 4; ++j) acc[i][j] = v8f{0.f,0.f,0.f,0.f,0.f,0.f,0.f,0.f};

  const int ktiles = K / GBK;
  for (int kt = 0; kt < ktiles; ++kt) {
    const int k0 = kt * GBK;
    // A tile: 128x32, row-major
    for (int c = tid; c < GBM * 4; c += 256) {
      const int row = c >> 2, ch = (c & 3) << 3;
      *(float4*)&As[row * GLD + ch] =
          *(const float4*)&A[(size_t)(m0 + row) * K + k0 + ch];
    }
    // B tile: 32x128, stored transposed Bs[n][k] (N tail guarded)
    for (int c = tid; c < GBK * (GBN / 8); c += 256) {
      const int k  = c >> 4;
      const int n8 = (c & 15) << 3;
      bf16_t tmp[8];
      if (n0 + n8 + 8 <= N) {
        *(float4*)tmp = *(const float4*)&B[(size_t)(k0 + k) * N + n0 + n8];
      } else {
#pragma unroll
        for (int j = 0; j < 8; ++j)
          tmp[j] = (n0 + n8 + j < N) ? B[(size_t)(k0 + k) * N + n0 + n8 + j]
                                     : (bf16_t)0.f;
      }
#pragma unroll
      for (int j = 0; j < 8; ++j) Bs[(n8 + j) * GLD + k] = tmp[j];
    }
    __syncthreads();

    v16bf af[2], bfr[4];
#pragma unroll
    for (int mf = 0; mf < 2; ++mf) af[mf]  = load_frag(As, GLD, wm + mf * 16, 0, lane);
#pragma unroll
    for (int nf = 0; nf < 4; ++nf) bfr[nf] = load_frag(Bs, GLD, wn + nf * 16, 0, lane);
#pragma unroll
    for (int mf = 0; mf < 2; ++mf)
#pragma unroll
      for (int nf = 0; nf < 4; ++nf)
        acc[mf][nf] = wmma_bf16(af[mf], bfr[nf], acc[mf][nf]);
    __syncthreads();
  }

  const int halfsel = (lane >> 4) << 3;  // C layout: row = i + 8*(lane>=16)
  const int ncol    = lane & 15;
#pragma unroll
  for (int mf = 0; mf < 2; ++mf)
#pragma unroll
    for (int nf = 0; nf < 4; ++nf)
#pragma unroll
      for (int i = 0; i < 8; ++i) {
        const int m = m0 + wm + mf * 16 + i + halfsel;
        const int n = n0 + wn + nf * 16 + ncol;
        if (n < N) {
          const size_t idx = (size_t)m * N + n;
          const float  v   = acc[mf][nf][i];
          if (mode == 0)      C[idx] = v;
          else if (mode == 1) C[idx] = v + resid[idx];
          else                C[idx] += rowscale[(size_t)m * rs_stride] * v;
        }
      }
}

// ---------------------------------------------------------------------------
// Fused MoE gate/up: G[M,N](bf16) = silu(X@Wg) * (X@Wu), optional tile skip
// ---------------------------------------------------------------------------
#define UBN 64
__global__ __launch_bounds__(256) void moe_gateup_k(
    const bf16_t* __restrict__ X, const bf16_t* __restrict__ Wg,
    const bf16_t* __restrict__ Wu, bf16_t* __restrict__ G,
    const float* __restrict__ rowscale, int rs_stride,
    int M, int N, int K)
{
  __shared__ alignas(16) bf16_t As[GBM * GLD];
  __shared__ alignas(16) bf16_t Bgs[UBN * GLD];
  __shared__ alignas(16) bf16_t Bus[UBN * GLD];
  __shared__ int s_live;

  const int tid  = threadIdx.x;
  const int lane = tid & 31;
  const int wave = tid >> 5;
  const int m0 = blockIdx.y * GBM;
  const int n0 = blockIdx.x * UBN;

  if (rowscale) {
    if (tid == 0) s_live = 0;
    __syncthreads();
    if (tid < GBM && rowscale[(size_t)(m0 + tid) * rs_stride] != 0.0f) s_live = 1;
    __syncthreads();
    if (!s_live) return;
  }

  const int wm = (wave & 3) * 32;
  const int wn = (wave >> 2) * 32;

  v8f accg[2][2], accu[2][2];
#pragma unroll
  for (int i = 0; i < 2; ++i)
#pragma unroll
    for (int j = 0; j < 2; ++j) {
      accg[i][j] = v8f{0.f,0.f,0.f,0.f,0.f,0.f,0.f,0.f};
      accu[i][j] = v8f{0.f,0.f,0.f,0.f,0.f,0.f,0.f,0.f};
    }

  const int ktiles = K / GBK;
  for (int kt = 0; kt < ktiles; ++kt) {
    const int k0 = kt * GBK;
    for (int c = tid; c < GBM * 4; c += 256) {
      const int row = c >> 2, ch = (c & 3) << 3;
      *(float4*)&As[row * GLD + ch] =
          *(const float4*)&X[(size_t)(m0 + row) * K + k0 + ch];
    }
    {
      const int k  = tid >> 3;              // 256 threads == 256 chunks
      const int n8 = (tid & 7) << 3;
      bf16_t tg[8], tu[8];
      *(float4*)tg = *(const float4*)&Wg[(size_t)(k0 + k) * N + n0 + n8];
      *(float4*)tu = *(const float4*)&Wu[(size_t)(k0 + k) * N + n0 + n8];
#pragma unroll
      for (int j = 0; j < 8; ++j) {
        Bgs[(n8 + j) * GLD + k] = tg[j];
        Bus[(n8 + j) * GLD + k] = tu[j];
      }
    }
    __syncthreads();

    v16bf af[2], bg[2], bu[2];
#pragma unroll
    for (int mf = 0; mf < 2; ++mf) af[mf] = load_frag(As, GLD, wm + mf * 16, 0, lane);
#pragma unroll
    for (int nf = 0; nf < 2; ++nf) {
      bg[nf] = load_frag(Bgs, GLD, wn + nf * 16, 0, lane);
      bu[nf] = load_frag(Bus, GLD, wn + nf * 16, 0, lane);
    }
#pragma unroll
    for (int mf = 0; mf < 2; ++mf)
#pragma unroll
      for (int nf = 0; nf < 2; ++nf) {
        accg[mf][nf] = wmma_bf16(af[mf], bg[nf], accg[mf][nf]);
        accu[mf][nf] = wmma_bf16(af[mf], bu[nf], accu[mf][nf]);
      }
    __syncthreads();
  }

  const int halfsel = (lane >> 4) << 3;
  const int ncol    = lane & 15;
#pragma unroll
  for (int mf = 0; mf < 2; ++mf)
#pragma unroll
    for (int nf = 0; nf < 2; ++nf)
#pragma unroll
      for (int i = 0; i < 8; ++i) {
        const int m = m0 + wm + mf * 16 + i + halfsel;
        const int n = n0 + wn + nf * 16 + ncol;
        const float g = accg[mf][nf][i];
        const float u = accu[mf][nf][i];
        const float s = g * (1.0f / (1.0f + __expf(-g))) * u;   // silu(g)*u
        G[(size_t)m * N + n] = (bf16_t)s;
      }
}

// ---------------------------------------------------------------------------
// Flash attention (causal): Q,K in [B,H,S,192] bf16, V in [B,H,S,128] bf16.
// 4 waves/block, 64 queries/block, 64-key tiles streamed via LDS.
// Out written bf16 as [B*S, H*128] for the o_proj GEMM.
// ---------------------------------------------------------------------------
#define FLD_K 200   // 192 + 8 pad
#define FLD_V 72    // 64  + 8 pad

__global__ __launch_bounds__(128) void flash_attn_k(
    const bf16_t* __restrict__ Qp, const bf16_t* __restrict__ Kp,
    const bf16_t* __restrict__ Vp, const int* __restrict__ amask,
    bf16_t* __restrict__ Out)
{
  __shared__ alignas(16) bf16_t Ks[64 * FLD_K];
  __shared__ alignas(16) bf16_t Vs[DVV * FLD_V];       // transposed [dv][key]
  __shared__ alignas(16) bf16_t Ps[4 * 16 * FLD_V];    // per-wave P scratch

  const int tid  = threadIdx.x;
  const int lane = tid & 31;
  const int wave = tid >> 5;
  const int qb = blockIdx.x, h = blockIdx.y, b = blockIdx.z;
  const int q0 = qb * 64;

  const bf16_t* Qhead = Qp + (size_t)(b * H16 + h) * SEQ * DQ;
  const bf16_t* Khead = Kp + (size_t)(b * H16 + h) * SEQ * DQ;
  const bf16_t* Vhead = Vp + (size_t)(b * H16 + h) * SEQ * DVV;

  // Q fragments straight from global (row-major layout == fragment layout)
  v16bf qf[6];
#pragma unroll
  for (int kf = 0; kf < 6; ++kf)
    qf[kf] = load_frag(Qhead + (size_t)(q0 + wave * 16) * DQ, DQ, 0, kf * 32, lane);

  v8f o[8];
  float rmax[8], rsum[8];
#pragma unroll
  for (int f = 0; f < 8; ++f) o[f] = v8f{0.f,0.f,0.f,0.f,0.f,0.f,0.f,0.f};
#pragma unroll
  for (int i = 0; i < 8; ++i) { rmax[i] = -3.0e38f; rsum[i] = 0.f; }

  const int rowbase = q0 + wave * 16 + ((lane >> 4) << 3);  // + i = query row
  const int colb    = lane & 15;
  const float scale = 0.07216878364870323f;                 // 192^-0.5

  for (int kb = 0; kb <= qb; ++kb) {
    const int k0 = kb * 64;
    for (int c = tid; c < 64 * 24; c += 128) {              // K tile 64x192
      const int r = c / 24, ch = (c % 24) << 3;
      *(float4*)&Ks[r * FLD_K + ch] =
          *(const float4*)&Khead[(size_t)(k0 + r) * DQ + ch];
    }
    for (int c = tid; c < 64 * 16; c += 128) {              // V tile transposed
      const int r = c >> 4, ch = (c & 15) << 3;
      bf16_t tmp[8];
      *(float4*)tmp = *(const float4*)&Vhead[(size_t)(k0 + r) * DVV + ch];
#pragma unroll
      for (int j = 0; j < 8; ++j) Vs[(ch + j) * FLD_V + r] = tmp[j];
    }
    __syncthreads();

    // S = Q K^T (16x64 per wave)
    v8f sacc[4];
#pragma unroll
    for (int nf = 0; nf < 4; ++nf) sacc[nf] = v8f{0.f,0.f,0.f,0.f,0.f,0.f,0.f,0.f};
#pragma unroll
    for (int kf = 0; kf < 6; ++kf)
#pragma unroll
      for (int nf = 0; nf < 4; ++nf) {
        v16bf kfr = load_frag(Ks, FLD_K, nf * 16, kf * 32, lane);
        sacc[nf] = wmma_bf16(qf[kf], kfr, sacc[nf]);
      }

    int am[4];
#pragma unroll
    for (int nf = 0; nf < 4; ++nf) am[nf] = amask[b * SEQ + k0 + nf * 16 + colb];

    // online softmax, row stats live across the 16-lane column groups
#pragma unroll
    for (int i = 0; i < 8; ++i) {
      const int row = rowbase + i;
      float mloc = -3.0e38f;
#pragma unroll
      for (int nf = 0; nf < 4; ++nf) {
        const int col = k0 + nf * 16 + colb;
        float s = sacc[nf][i] * scale;
        s = ((col <= row) && (am[nf] > 0)) ? s : -3.0e38f;
        sacc[nf][i] = s;
        mloc = fmaxf(mloc, s);
      }
      mloc = fmaxf(mloc, __shfl_xor(mloc, 1));
      mloc = fmaxf(mloc, __shfl_xor(mloc, 2));
      mloc = fmaxf(mloc, __shfl_xor(mloc, 4));
      mloc = fmaxf(mloc, __shfl_xor(mloc, 8));
      const float nm   = fmaxf(rmax[i], mloc);
      const float corr = __expf(rmax[i] - nm);
      rmax[i] = nm;
      rsum[i] *= corr;
#pragma unroll
      for (int f = 0; f < 8; ++f) o[f][i] *= corr;
      float ps = 0.f;
#pragma unroll
      for (int nf = 0; nf < 4; ++nf) {
        const float p = __expf(sacc[nf][i] - nm);
        ps += p;
        Ps[(wave * 16 + i + ((lane >> 4) << 3)) * FLD_V + nf * 16 + colb] = (bf16_t)p;
      }
      ps += __shfl_xor(ps, 1);
      ps += __shfl_xor(ps, 2);
      ps += __shfl_xor(ps, 4);
      ps += __shfl_xor(ps, 8);
      rsum[i] += ps;
    }

    // our own wave's DS stores must land before re-reading P as an A fragment
    asm volatile("s_wait_dscnt 0x0" ::: "memory");

    v16bf pf[2];
#pragma unroll
    for (int kf = 0; kf < 2; ++kf)
      pf[kf] = load_frag(Ps + wave * 16 * FLD_V, FLD_V, 0, kf * 32, lane);
#pragma unroll
    for (int nf = 0; nf < 8; ++nf)
#pragma unroll
      for (int kf = 0; kf < 2; ++kf) {
        v16bf vfr = load_frag(Vs, FLD_V, nf * 16, kf * 32, lane);
        o[nf] = wmma_bf16(pf[kf], vfr, o[nf]);
      }
    __syncthreads();
  }

#pragma unroll
  for (int nf = 0; nf < 8; ++nf)
#pragma unroll
    for (int i = 0; i < 8; ++i) {
      const int row = rowbase + i;
      const float v = o[nf][i] / rsum[i];
      Out[(size_t)(b * SEQ + row) * (H16 * DVV) + h * DVV + nf * 16 + colb] =
          (bf16_t)v;
    }
}

// ---------------------------------------------------------------------------
// RMSNorm over rows (optional f32 and/or bf16 outputs)
// ---------------------------------------------------------------------------
__global__ __launch_bounds__(256) void rmsnorm_k(
    const float* __restrict__ X, const float* __restrict__ W,
    float* __restrict__ Yf, bf16_t* __restrict__ Yb,
    int N, int inStride, int outStride)
{
  const int row = blockIdx.x;
  const int tid = threadIdx.x, lane = tid & 31, wave = tid >> 5;
  const float* x = X + (size_t)row * inStride;
  float ss = 0.f;
  for (int i = tid; i < N; i += 256) { const float v = x[i]; ss += v * v; }
  ss += __shfl_xor(ss, 1);
  ss += __shfl_xor(ss, 2);
  ss += __shfl_xor(ss, 4);
  ss += __shfl_xor(ss, 8);
  ss += __shfl_xor(ss, 16);
  __shared__ float red[8];
  __shared__ float s_inv;
  if (lane == 0) red[wave] = ss;
  __syncthreads();
  if (tid == 0) {
    float t = 0.f;
    for (int k = 0; k < 8; ++k) t += red[k];
    s_inv = rsqrtf(t / (float)N + 1e-6f);
  }
  __syncthreads();
  const float inv = s_inv;
  for (int i = tid; i < N; i += 256) {
    const float v = x[i] * inv * W[i];
    if (Yf) Yf[(size_t)row * outStride + i] = v;
    if (Yb) Yb[(size_t)row * outStride + i] = (bf16_t)v;
  }
}

// ---------------------------------------------------------------------------
// RoPE (interleave-to-half) + pack q into [B,H,S,192] bf16
// ---------------------------------------------------------------------------
__global__ __launch_bounds__(256) void rope_pack_q_k(
    const float* __restrict__ q, const float* __restrict__ cosT,
    const float* __restrict__ sinT, const int* __restrict__ pos,
    bf16_t* __restrict__ Qp)
{
  const int idx = blockIdx.x * blockDim.x + threadIdx.x;
  if (idx >= NTOK * H16) return;
  const int h = idx & 15;
  const int t = idx >> 4;
  const int b = t / SEQ, s = t - b * SEQ;
  const float* qp = q + ((size_t)t * H16 + h) * DQ;
  bf16_t* op = Qp + ((size_t)(b * H16 + h) * SEQ + s) * DQ;
  const int p = pos[t];
  const float* cs = cosT + (size_t)p * DR;
  const float* sn = sinT + (size_t)p * DR;
  for (int d = 0; d < DN; ++d) op[d] = (bf16_t)qp[d];
  float xi[DR];
#pragma unroll
  for (int j = 0; j < 32; ++j) { xi[j] = qp[DN + 2 * j]; xi[j + 32] = qp[DN + 2 * j + 1]; }
#pragma unroll
  for (int j = 0; j < 32; ++j) {
    op[DN + j]      = (bf16_t)(xi[j]      * cs[j]      - xi[j + 32] * sn[j]);
    op[DN + 32 + j] = (bf16_t)(xi[j + 32] * cs[32 + j] + xi[j]      * sn[32 + j]);
  }
}

// ---------------------------------------------------------------------------
// Pack K (k_nope + broadcast roped k_pe) and V from kv_b output
// ---------------------------------------------------------------------------
__global__ __launch_bounds__(256) void pack_kv_k(
    const float* __restrict__ kv, const float* __restrict__ ckv,
    const float* __restrict__ cosT, const float* __restrict__ sinT,
    const int* __restrict__ pos, bf16_t* __restrict__ Kp, bf16_t* __restrict__ Vp)
{
  const int idx = blockIdx.x * blockDim.x + threadIdx.x;
  if (idx >= NTOK * H16) return;
  const int h = idx & 15;
  const int t = idx >> 4;
  const int b = t / SEQ, s = t - b * SEQ;
  const float* kvp = kv + ((size_t)t * H16 + h) * (DN + DVV);
  bf16_t* kop = Kp + ((size_t)(b * H16 + h) * SEQ + s) * DQ;
  bf16_t* vop = Vp + ((size_t)(b * H16 + h) * SEQ + s) * DVV;
  for (int d = 0; d < DN; ++d) {
    kop[d] = (bf16_t)kvp[d];
    vop[d] = (bf16_t)kvp[DN + d];
  }
  const float* pe = ckv + (size_t)t * (KVL + DR) + KVL;
  const int p = pos[t];
  const float* cs = cosT + (size_t)p * DR;
  const float* sn = sinT + (size_t)p * DR;
  float xi[DR];
#pragma unroll
  for (int j = 0; j < 32; ++j) { xi[j] = pe[2 * j]; xi[j + 32] = pe[2 * j + 1]; }
#pragma unroll
  for (int j = 0; j < 32; ++j) {
    kop[DN + j]      = (bf16_t)(xi[j]      * cs[j]      - xi[j + 32] * sn[j]);
    kop[DN + 32 + j] = (bf16_t)(xi[j + 32] * cs[32 + j] + xi[j]      * sn[32 + j]);
  }
}

// ---------------------------------------------------------------------------
// MoE router: softmax over 8 experts, top-2 combine weights per token
// ---------------------------------------------------------------------------
__global__ __launch_bounds__(32) void gate_k(
    const bf16_t* __restrict__ X, const float* __restrict__ GW,
    float* __restrict__ comb)
{
  const int t = blockIdx.x;
  const int lane = threadIdx.x;
  float s[NEXP];
#pragma unroll
  for (int e = 0; e < NEXP; ++e) s[e] = 0.f;
  for (int d = lane; d < HIDDEN; d += 32) {
    const float x = (float)X[(size_t)t * HIDDEN + d];
#pragma unroll
    for (int e = 0; e < NEXP; ++e) s[e] += x * GW[d * NEXP + e];
  }
#pragma unroll
  for (int e = 0; e < NEXP; ++e) {
    s[e] += __shfl_xor(s[e], 1);
    s[e] += __shfl_xor(s[e], 2);
    s[e] += __shfl_xor(s[e], 4);
    s[e] += __shfl_xor(s[e], 8);
    s[e] += __shfl_xor(s[e], 16);
  }
  if (lane == 0) {
    float m = s[0];
    for (int e = 1; e < NEXP; ++e) m = fmaxf(m, s[e]);
    float p[NEXP], psum = 0.f;
    for (int e = 0; e < NEXP; ++e) { p[e] = __expf(s[e] - m); psum += p[e]; }
    for (int e = 0; e < NEXP; ++e) p[e] /= psum;
    int i1 = 0;
    for (int e = 1; e < NEXP; ++e) if (p[e] > p[i1]) i1 = e;
    int i2 = -1;
    for (int e = 0; e < NEXP; ++e)
      if (e != i1 && (i2 < 0 || p[e] > p[i2])) i2 = e;
    for (int e = 0; e < NEXP; ++e)
      comb[(size_t)t * NEXP + e] = (e == i1) ? p[i1] : (e == i2) ? p[i2] : 0.f;
  }
}

// ---------------------------------------------------------------------------
// Elementwise helpers
// ---------------------------------------------------------------------------
__global__ __launch_bounds__(256) void cvt_bf16_k(const float* __restrict__ x,
                                                  bf16_t* __restrict__ y, size_t n) {
  size_t i = (size_t)blockIdx.x * blockDim.x + threadIdx.x;
  const size_t stride = (size_t)gridDim.x * blockDim.x;
  for (; i < n; i += stride) y[i] = (bf16_t)x[i];
}

__global__ __launch_bounds__(256) void add_k(const float* __restrict__ a,
                                             const float* __restrict__ b,
                                             float* __restrict__ o, size_t n) {
  size_t i = (size_t)blockIdx.x * blockDim.x + threadIdx.x;
  const size_t stride = (size_t)gridDim.x * blockDim.x;
  for (; i < n; i += stride) o[i] = a[i] + b[i];
}

// ---------------------------------------------------------------------------
// Host orchestration
// ---------------------------------------------------------------------------
extern "C" void kernel_launch(void* const* d_in, const int* in_sizes, int n_in,
                              void* d_out, int out_size, void* d_ws, size_t ws_size,
                              hipStream_t stream)
{
  (void)in_sizes; (void)n_in; (void)out_size; (void)ws_size;
  const float* hidden    = (const float*)d_in[0];
  const int*   amask     = (const int*)d_in[1];
  const int*   posids    = (const int*)d_in[2];
  const float* ropeC     = (const float*)d_in[3];
  const float* ropeS     = (const float*)d_in[4];
  const float* w_in_ln   = (const float*)d_in[5];
  const float* w_q_a     = (const float*)d_in[6];
  const float* w_q_a_ln  = (const float*)d_in[7];
  const float* w_q_b     = (const float*)d_in[8];
  const float* w_kv_a    = (const float*)d_in[9];
  const float* w_kv_a_ln = (const float*)d_in[10];
  const float* w_kv_b    = (const float*)d_in[11];
  const float* w_o       = (const float*)d_in[12];
  const float* w_post_ln = (const float*)d_in[13];
  const float* gate_w    = (const float*)d_in[14];
  const float* eg        = (const float*)d_in[15];
  const float* eu        = (const float*)d_in[16];
  const float* ed        = (const float*)d_in[17];
  const float* sg        = (const float*)d_in[18];
  const float* su        = (const float*)d_in[19];
  const float* sd        = (const float*)d_in[20];
  float* out = (float*)d_out;

  char* wsb = (char*)d_ws;
  size_t off = 0;
  auto alloc = [&](size_t bytes) -> char* {
    char* p = wsb + off;
    off += (bytes + 255) & ~(size_t)255;
    return p;
  };

  // bf16 weight copies (expert weights ~130MB bf16 -> resident in 192MB L2)
  bf16_t* wqa_b  = (bf16_t*)alloc((size_t)HIDDEN * QLORA_ * 2);
  bf16_t* wqb_b  = (bf16_t*)alloc((size_t)QLORA_ * (H16 * DQ) * 2);
  bf16_t* wkva_b = (bf16_t*)alloc((size_t)HIDDEN * (KVL + DR) * 2);
  bf16_t* wkvb_b = (bf16_t*)alloc((size_t)KVL * (H16 * (DN + DVV)) * 2);
  bf16_t* wo_b   = (bf16_t*)alloc((size_t)(H16 * DVV) * HIDDEN * 2);
  bf16_t* eg_b   = (bf16_t*)alloc((size_t)NEXP * HIDDEN * INTER_ * 2);
  bf16_t* eu_b   = (bf16_t*)alloc((size_t)NEXP * HIDDEN * INTER_ * 2);
  bf16_t* ed_b   = (bf16_t*)alloc((size_t)NEXP * INTER_ * HIDDEN * 2);
  bf16_t* sg_b   = (bf16_t*)alloc((size_t)HIDDEN * SINTER * 2);
  bf16_t* su_b   = (bf16_t*)alloc((size_t)HIDDEN * SINTER * 2);
  bf16_t* sd_b   = (bf16_t*)alloc((size_t)SINTER * HIDDEN * 2);

  // activations
  bf16_t* hnorm  = (bf16_t*)alloc((size_t)NTOK * HIDDEN * 2);
  float*  qa     = (float*) alloc((size_t)NTOK * QLORA_ * 4);
  bf16_t* qan    = (bf16_t*)alloc((size_t)NTOK * QLORA_ * 2);
  float*  qf     = (float*) alloc((size_t)NTOK * H16 * DQ * 4);
  bf16_t* Qpack  = (bf16_t*)alloc((size_t)NTOK * H16 * DQ * 2);
  float*  ckv    = (float*) alloc((size_t)NTOK * (KVL + DR) * 4);
  bf16_t* kvn    = (bf16_t*)alloc((size_t)NTOK * KVL * 2);
  float*  kvout  = (float*) alloc((size_t)NTOK * H16 * (DN + DVV) * 4);
  bf16_t* Kpack  = (bf16_t*)alloc((size_t)NTOK * H16 * DQ * 2);
  bf16_t* Vpack  = (bf16_t*)alloc((size_t)NTOK * H16 * DVV * 2);
  bf16_t* attnb  = (bf16_t*)alloc((size_t)NTOK * H16 * DVV * 2);
  float*  h2     = (float*) alloc((size_t)NTOK * HIDDEN * 4);
  bf16_t* xmoe   = (bf16_t*)alloc((size_t)NTOK * HIDDEN * 2);
  float*  comb   = (float*) alloc((size_t)NTOK * NEXP * 4);
  bf16_t* gbuf   = (bf16_t*)alloc((size_t)NTOK * SINTER * 2);
  float*  moe    = (float*) alloc((size_t)NTOK * HIDDEN * 4);

  auto cvt = [&](const float* x, bf16_t* y, size_t n) {
    cvt_bf16_k<<<2048, 256, 0, stream>>>(x, y, n);
  };
  auto gemm = [&](const bf16_t* A, const bf16_t* B, float* C, const float* resid,
                  const float* rowscale, int rs_stride, int M, int N, int K, int mode) {
    dim3 g((N + GBN - 1) / GBN, M / GBM);
    gemm_bf16_k<<<g, 256, 0, stream>>>(A, B, C, resid, rowscale, rs_stride,
                                       M, N, K, mode);
  };

  // --- weight conversion (one pass; read fp32 once, write bf16 once) ---
  cvt(w_q_a,  wqa_b,  (size_t)HIDDEN * QLORA_);
  cvt(w_q_b,  wqb_b,  (size_t)QLORA_ * (H16 * DQ));
  cvt(w_kv_a, wkva_b, (size_t)HIDDEN * (KVL + DR));
  cvt(w_kv_b, wkvb_b, (size_t)KVL * (H16 * (DN + DVV)));
  cvt(w_o,    wo_b,   (size_t)(H16 * DVV) * HIDDEN);
  cvt(eg, eg_b, (size_t)NEXP * HIDDEN * INTER_);
  cvt(eu, eu_b, (size_t)NEXP * HIDDEN * INTER_);
  cvt(ed, ed_b, (size_t)NEXP * INTER_ * HIDDEN);
  cvt(sg, sg_b, (size_t)HIDDEN * SINTER);
  cvt(su, su_b, (size_t)HIDDEN * SINTER);
  cvt(sd, sd_b, (size_t)SINTER * HIDDEN);

  // --- attention ---
  rmsnorm_k<<<NTOK, 256, 0, stream>>>(hidden, w_in_ln, nullptr, hnorm,
                                      HIDDEN, HIDDEN, HIDDEN);
  gemm(hnorm, wqa_b, qa, nullptr, nullptr, 0, NTOK, QLORA_, HIDDEN, 0);
  rmsnorm_k<<<NTOK, 256, 0, stream>>>(qa, w_q_a_ln, nullptr, qan,
                                      QLORA_, QLORA_, QLORA_);
  gemm(qan, wqb_b, qf, nullptr, nullptr, 0, NTOK, H16 * DQ, QLORA_, 0);
  gemm(hnorm, wkva_b, ckv, nullptr, nullptr, 0, NTOK, KVL + DR, HIDDEN, 0);
  rmsnorm_k<<<NTOK, 256, 0, stream>>>(ckv, w_kv_a_ln, nullptr, kvn,
                                      KVL, KVL + DR, KVL);
  gemm(kvn, wkvb_b, kvout, nullptr, nullptr, 0, NTOK, H16 * (DN + DVV), KVL, 0);

  rope_pack_q_k<<<(NTOK * H16 + 255) / 256, 256, 0, stream>>>(qf, ropeC, ropeS,
                                                              posids, Qpack);
  pack_kv_k<<<(NTOK * H16 + 255) / 256, 256, 0, stream>>>(kvout, ckv, ropeC,
                                                          ropeS, posids, Kpack,
                                                          Vpack);

  flash_attn_k<<<dim3(SEQ / 64, H16, BATCH), 128, 0, stream>>>(Qpack, Kpack,
                                                               Vpack, amask,
                                                               attnb);
  // o_proj + residual
  gemm(attnb, wo_b, h2, hidden, nullptr, 0, NTOK, HIDDEN, H16 * DVV, 1);

  // --- MoE ---
  rmsnorm_k<<<NTOK, 256, 0, stream>>>(h2, w_post_ln, nullptr, xmoe,
                                      HIDDEN, HIDDEN, HIDDEN);
  gate_k<<<NTOK, 32, 0, stream>>>(xmoe, gate_w, comb);

  // shared expert -> moe
  moe_gateup_k<<<dim3(SINTER / UBN, NTOK / GBM), 256, 0, stream>>>(
      xmoe, sg_b, su_b, gbuf, nullptr, 0, NTOK, SINTER, HIDDEN);
  gemm(gbuf, sd_b, moe, nullptr, nullptr, 0, NTOK, HIDDEN, SINTER, 0);

  // routed experts accumulate (tile-skipped by combine weights)
  for (int e = 0; e < NEXP; ++e) {
    const bf16_t* weg = eg_b + (size_t)e * HIDDEN * INTER_;
    const bf16_t* weu = eu_b + (size_t)e * HIDDEN * INTER_;
    const bf16_t* wed = ed_b + (size_t)e * INTER_ * HIDDEN;
    moe_gateup_k<<<dim3(INTER_ / UBN, NTOK / GBM), 256, 0, stream>>>(
        xmoe, weg, weu, gbuf, comb + e, NEXP, NTOK, INTER_, HIDDEN);
    gemm(gbuf, wed, moe, nullptr, comb + e, NEXP, NTOK, HIDDEN, INTER_, 2);
  }

  // final residual
  add_k<<<2048, 256, 0, stream>>>(h2, moe, out, (size_t)NTOK * HIDDEN);
}
